// GCNLayer_32993938767997
// MI455X (gfx1250) — matrix operand order repaired and checked by reading, not compiled, not run on hardware
//
#include <hip/hip_runtime.h>

// Problem constants (from reference): B,N,K,D,H
#define Bn 4
#define Nn 16384
#define Kn 32
#define Dn 128
#define Hn 128

#define MTILE    64      // rows per block (4 waves x 16-row WMMA tiles)
#define NTHREADS 128     // 4 wave32 waves
#define PW  136          // LDS pitch (bf16 elems) for Wt/Bwt: 272B, 16B-aligned rows
#define PA  136          // LDS pitch for agg/vf tiles

typedef __attribute__((ext_vector_type(16))) __bf16 v16bf;
typedef __attribute__((ext_vector_type(8)))  float  v8f;

union FragU { v16bf v; uint4 q[2]; };

// round-to-nearest-even f32 -> bf16
__device__ __forceinline__ unsigned short f2bf(float x) {
  unsigned int u = __float_as_uint(x);
  u += 0x7FFFu + ((u >> 16) & 1u);
  return (unsigned short)(u >> 16);
}

__global__ __launch_bounds__(NTHREADS)
void gcn_fused_wmma(const float* __restrict__ vf,
                    const int*   __restrict__ nidx,
                    const int*   __restrict__ lens,
                    const float* __restrict__ W,
                    const float* __restrict__ Bw,
                    float*       __restrict__ out) {
  __shared__ unsigned short sWt [Hn * PW];     // W^T  as bf16: sWt [col*PW + k]
  __shared__ unsigned short sBwt[Hn * PW];     // Bw^T as bf16
  __shared__ unsigned short sAgg[MTILE * PA];  // aggregated rows, bf16
  __shared__ unsigned short sVf [MTILE * PA];  // own rows, bf16

  const int tid  = threadIdx.x;
  const int lane = tid & 31;
  const int wave = tid >> 5;

  const int row0 = blockIdx.x * MTILE;          // flattened (b*N+n) row base
  const int b    = row0 / Nn;                   // MTILE divides N -> no straddle
  const float* vfb = vf + (size_t)b * Nn * Dn;  // per-graph gather base

  // ---- stage W^T and Bw^T into LDS as bf16 (coalesced 128B/wave reads) ----
  {
    const int col = wave * 32 + lane;           // 0..127
    #pragma unroll 4
    for (int k = 0; k < Dn; ++k) {
      sWt [col * PW + k] = f2bf(W [k * Hn + col]);
      sBwt[col * PW + k] = f2bf(Bw[k * Hn + col]);
    }
  }

  // ---- aggregate neighbors: one wave per 16-row tile ----------------------
  {
    const int rloc0 = wave * 16;
    for (int r = 0; r < 16; ++r) {
      const int rl  = rloc0 + r;
      const int row = row0 + rl;
      // K == 32 == wave width: lane l holds neighbor index k=l for this row
      const int myi = nidx[(size_t)row * Kn + lane];
      const int len = lens[row];
      float4 acc = make_float4(0.f, 0.f, 0.f, 0.f);
      for (int k = 0; k < len; ++k) {           // uniform trip count per wave
        const int j = __shfl(myi, k, 32);
        const float4 v = *(const float4*)(vfb + (size_t)j * Dn + (lane << 2));
        acc.x += v.x; acc.y += v.y; acc.z += v.z; acc.w += v.w;
      }
      const float s = 1.0f / (float)(len > 0 ? len : 1);
      uint2 p;
      p.x = (unsigned)f2bf(acc.x * s) | ((unsigned)f2bf(acc.y * s) << 16);
      p.y = (unsigned)f2bf(acc.z * s) | ((unsigned)f2bf(acc.w * s) << 16);
      *(uint2*)&sAgg[rl * PA + (lane << 2)] = p;

      const float4 own = *(const float4*)(vf + (size_t)row * Dn + (lane << 2));
      uint2 q;
      q.x = (unsigned)f2bf(own.x) | ((unsigned)f2bf(own.y) << 16);
      q.y = (unsigned)f2bf(own.z) | ((unsigned)f2bf(own.w) << 16);
      *(uint2*)&sVf[rl * PA + (lane << 2)] = q;
    }
  }

  __syncthreads();

  // ---- GEMM: out[16x128] = relu(agg@W + vf@Bw), one 16-row tile per wave --
  // A 16x32 bf16 fragment: lanes 0-15 = rows, K {kb+0..7, kb+16..23};
  //                        lanes 16-31 same rows, K {kb+8..15, kb+24..31}.
// B 32x16 bf16 fragment: lanes 0-15 = cols, K {kb+0..7, kb+16..23};
//                        lanes 16-31 same cols, K {kb+8..15, kb+24..31}.
  const int rloc0 = wave * 16;
  const int arow  = rloc0 + (lane & 15);
  const int koffA = (lane < 16) ? 0 : 8;
  const int koffB = (lane < 16) ? 0 : 8;
  const int rbase = row0 + rloc0 + ((lane < 16) ? 0 : 8);

  for (int nt = 0; nt < 8; ++nt) {
    const int col = nt * 16 + (lane & 15);
    v8f c = {0.f, 0.f, 0.f, 0.f, 0.f, 0.f, 0.f, 0.f};

    #pragma unroll
    for (int kb = 0; kb < Dn; kb += 32) {       // agg @ W
      FragU a, bf;
      a.q[0]  = *(const uint4*)&sAgg[arow * PA + kb +      koffA];
      a.q[1]  = *(const uint4*)&sAgg[arow * PA + kb + 16 + koffA];
      bf.q[0] = *(const uint4*)&sWt [col  * PW + kb + koffB];
      bf.q[1] = *(const uint4*)&sWt [col  * PW + kb + koffB + 16];
      c = __builtin_amdgcn_wmma_f32_16x16x32_bf16(false, a.v, false, bf.v,
                                                  (short)0, c, false, false);
    }
    #pragma unroll
    for (int kb = 0; kb < Dn; kb += 32) {       // vf @ Bw (same accumulator)
      FragU a, bf;
      a.q[0]  = *(const uint4*)&sVf [arow * PA + kb +      koffA];
      a.q[1]  = *(const uint4*)&sVf [arow * PA + kb + 16 + koffA];
      bf.q[0] = *(const uint4*)&sBwt[col  * PW + kb + koffB];
      bf.q[1] = *(const uint4*)&sBwt[col  * PW + kb + koffB + 16];
      c = __builtin_amdgcn_wmma_f32_16x16x32_bf16(false, a.v, false, bf.v,
                                                  (short)0, c, false, false);
    }

    // C/D layout: VGPR i -> row rbase+i (lanes 0-15) / rbase+i (+8 folded in)
    #pragma unroll
    for (int i = 0; i < 8; ++i) {
      const float v = c[i] > 0.f ? c[i] : 0.f;  // ReLU
      out[(size_t)(rbase + i) * Hn + col] = v;
    }
  }
}

extern "C" void kernel_launch(void* const* d_in, const int* in_sizes, int n_in,
                              void* d_out, int out_size, void* d_ws, size_t ws_size,
                              hipStream_t stream) {
  const float* vf   = (const float*)d_in[0];   // vertex_feat [B,N,D] f32
  const int*   nidx = (const int*)  d_in[1];   // neighbors_idx [B,N,K] i32
  const int*   lens = (const int*)  d_in[2];   // valid_lens [B,N] i32
  const float* W    = (const float*)d_in[3];   // W  [D,H] f32
  const float* Bw   = (const float*)d_in[4];   // Bw [D,H] f32
  float* out = (float*)d_out;                  // [B,N,H] f32

  dim3 grid((Bn * Nn) / MTILE);
  dim3 block(NTHREADS);
  hipLaunchKernelGGL(gcn_fused_wmma, grid, block, 0, stream,
                     vf, nidx, lens, W, Bw, out);
}
